// CrossAttention_57466662420776
// MI455X (gfx1250) — compile-verified, matrix operand order checked by reading, not run on hardware
//
#include <hip/hip_runtime.h>

// ---------------------------------------------------------------------------
// CrossAttention for MI455X (gfx1250, wave32, WMMA, async-to-LDS).
//
// Roofline: ~22 GFLOP total.  Naive path writes 512 MB of NxN scores to HBM
// (~44us @ 23.3 TB/s); fused flash-attention keeps scores in registers/LDS.
// Compute then dominates -> all GEMMs use v_wmma_f32_16x16x32_f16.
//
// Attention data movement: all 8 waves of a workgroup sweep the SAME K/V
// stream, so K/V blocks are staged into LDS once per workgroup using
// CDNA5 GLOBAL_LOAD_ASYNC_TO_LDS_B128 (ASYNCcnt), double-buffered so the
// next block's DMA overlaps the current block's WMMA + online softmax.
//
// Pipeline:
//   1) proj_qkv_kernel x3 : q/k/v = in @ W^T + b  -> f16 in workspace.
//      q,k stored [T,H,N,32]; v stored TRANSPOSED [T,H,32,N] so the P@V
//      B-operand is contiguous along the K (key) dimension.
//   2) attn_kernel        : flash attention per (t,h); async K/V staging.
//   3) out_proj_kernel    : out = attn @ Wo^T + bo (f32 output).
// ---------------------------------------------------------------------------

typedef __attribute__((ext_vector_type(16))) _Float16 v16h;
typedef __attribute__((ext_vector_type(8)))  _Float16 v8h;
typedef __attribute__((ext_vector_type(8)))  float    v8f;

#define HIDDEN 256
#define HEADS  8
#define HD     32
#define NTOK   2048
#define TSTEPS 4
#define ROWS   (TSTEPS * NTOK)          /* 8192 total tokens */
#define SCALE  0.17677669529663687f     /* 32^-0.5 */

// A/B operand layout for v_wmma_f32_16x16x32_f16 (ISA 7.12.2, 16-bit A 16x32):
// lanes 0-15 own row M=lane with K elems {0..7, 16..23};
// lanes 16-31 own row M=lane-16 with K elems {8..15, 24..31}.
// lo = (lane<16) ? 0 : 8 selects the half-wave K offset.

static __device__ inline v16h load_ab_f32(const float* __restrict__ base,
                                          int kk, int lo) {
  v16h r;
#pragma unroll
  for (int e = 0; e < 8; ++e) {
    r[e]     = (_Float16)base[kk + lo + e];
    r[e + 8] = (_Float16)base[kk + 16 + lo + e];
  }
  return r;
}

static __device__ inline v16h load_ab_f16(const _Float16* base, int kk, int lo) {
  v8h a = *(const v8h*)(base + kk + lo);        // 16B -> ds/global_load_b128
  v8h b = *(const v8h*)(base + kk + 16 + lo);
  v16h r;
#pragma unroll
  for (int e = 0; e < 8; ++e) { r[e] = a[e]; r[e + 8] = b[e]; }
  return r;
}

static __device__ inline v8f wmma16(v16h a, v16h b, v8f c) {
  return __builtin_amdgcn_wmma_f32_16x16x32_f16(
      /*neg_a=*/false, a, /*neg_b=*/false, b,
      /*c_mod=*/(short)0, c, /*reuse_a=*/false, /*reuse_b=*/false);
}

// CDNA5 async copy: global -> LDS, 16 bytes, tracked by ASYNCcnt.
// VDST carries the LDS byte address (low 32 bits of the flat shared address
// == LDS offset per the aperture rules), VADDR the 64-bit global address.
static __device__ inline void async_copy_b128(_Float16* lds_dst,
                                              const _Float16* gsrc) {
  unsigned int loff = (unsigned int)(unsigned long long)lds_dst;
  asm volatile("global_load_async_to_lds_b128 %0, %1, off"
               :: "v"(loff), "v"(gsrc) : "memory");
}

static __device__ inline void wait_asynccnt0() {
  asm volatile("s_wait_asynccnt 0x0" ::: "memory");
}

// ---------------------------------------------------------------------------
// Kernel 1: Y = X @ W^T + b, f32 in, f16 out (per-head layouts).
// One wave owns one 16x16 output tile; 8 waves/block; 8192 tiles total.
// vmode=0: out[((t*H+h)*N + n)*32 + d]   (q, k)
// vmode=1: out[((t*H+h)*32 + d)*N + n]   (v transposed)
// ---------------------------------------------------------------------------
__global__ __launch_bounds__(256) void proj_qkv_kernel(
    const float* __restrict__ X, const float* __restrict__ W,
    const float* __restrict__ bias, _Float16* __restrict__ out, int vmode) {
  const int lane = threadIdx.x & 31;
  const int wave = threadIdx.x >> 5;
  const int tid  = blockIdx.x * 8 + wave;   // wave-tile id, 0..8191
  const int row_tile = tid >> 4;            // 512 row tiles of 16
  const int col_tile = tid & 15;            // 16  col tiles of 16
  const int lo = (lane < 16) ? 0 : 8;
  const int nn = lane & 15;
  const int hi = lane >> 4;

  const float* xrow = X + (size_t)(row_tile * 16 + nn) * HIDDEN; // A row/lane
  const float* wrow = W + (size_t)(col_tile * 16 + nn) * HIDDEN; // B col = W row

  v8f acc = {};
#pragma unroll
  for (int kk = 0; kk < HIDDEN; kk += 32) {
    v16h a = load_ab_f32(xrow, kk, lo);
    v16h b = load_ab_f32(wrow, kk, lo);
    acc = wmma16(a, b, acc);
  }

  const int col = col_tile * 16 + nn;
  const float bv = bias[col];
  const int h = col >> 5, d = col & (HD - 1);
#pragma unroll
  for (int r = 0; r < 8; ++r) {
    const int row = row_tile * 16 + r + 8 * hi;   // C-layout: M = r + 8*half
    const int t = row >> 11, nl = row & (NTOK - 1);
    const float y = acc[r] + bv;
    const size_t idx = vmode
        ? ((size_t)((t * HEADS + h) * HD + d)) * NTOK + nl
        : ((size_t)((t * HEADS + h) * NTOK + nl)) * HD + d;
    out[idx] = (_Float16)y;
  }
}

// ---------------------------------------------------------------------------
// Kernel 2: flash attention per (t,h).  Grid = 32 heads * 16 blocks of 128
// query rows; 8 waves/block, 1 wave = one 16-row Q tile (hd=32 = one WMMA
// A operand).  K/V staged through double-buffered LDS via async DMA.
// ---------------------------------------------------------------------------
__global__ __launch_bounds__(256) void attn_kernel(
    const _Float16* __restrict__ qh, const _Float16* __restrict__ kh,
    const _Float16* __restrict__ vt, _Float16* __restrict__ aout) {
  __shared__ _Float16 kbuf[2][32][32];        // 32 keys x 32 hd   (2 KB/stage)
  __shared__ _Float16 vbuf[2][32][32];        // 32 hd   x 32 keys (2 KB/stage)
  __shared__ _Float16 pbuf_all[8][16 * 32];   // per-wave P tile   (8 KB)

  const int tid  = threadIdx.x;
  const int lane = tid & 31;
  const int wave = tid >> 5;
  const int th   = blockIdx.x >> 4;           // t*H + h, 0..31
  const int blk  = blockIdx.x & 15;           // 128-row query block
  const int t = th >> 3, h = th & 7;
  const int q0 = blk * 128 + wave * 16;
  const int lo = (lane < 16) ? 0 : 8;
  const int nn = lane & 15;
  const int hi = lane >> 4;

  const _Float16* qhead = qh + (size_t)th * NTOK * HD;
  const _Float16* khead = kh + (size_t)th * NTOK * HD;
  const _Float16* vhead = vt + (size_t)th * HD * NTOK;   // [32][2048]
  _Float16* pbuf = pbuf_all[wave];

  // Cooperative stage: 256 threads x one b128 each = 2 KB K + 2 KB V.
  const int krow  = tid >> 2;            // tid<128 : key row 0..31
  const int vrow  = (tid - 128) >> 2;    // tid>=128: hd row  0..31
  const int chunk = (tid & 3) * 8;       // 4 x 8 f16 chunks per 64 B row

  // Q tile held in registers for the whole K sweep.
  const v16h qa = load_ab_f16(qhead + (size_t)(q0 + nn) * HD, 0, lo);

  v8f o0 = {}, o1 = {};               // output accum, hd cols 0-15 / 16-31
  float mi[8], li[8];                 // row max / row sum, row = r + 8*hi
#pragma unroll
  for (int r = 0; r < 8; ++r) { mi[r] = -1e30f; li[r] = 0.f; }

  // Prologue: async-stage block 0 into stage 0.
  if (tid < 128) async_copy_b128(&kbuf[0][krow][chunk],
                                 khead + (size_t)krow * HD + chunk);
  else           async_copy_b128(&vbuf[0][vrow][chunk],
                                 vhead + (size_t)vrow * NTOK + chunk);

  const int nblk = NTOK / 32;           // 64 key blocks
  for (int i = 0; i < nblk; ++i) {
    const int p = i & 1;
    wait_asynccnt0();                   // my stage-p DMA complete
    __syncthreads();                    // everyone's DMA landed; prev reads done

    // Overlap: kick off next block's DMA into the other stage.
    if (i + 1 < nblk) {
      const int kb2 = (i + 1) * 32;
      if (tid < 128) async_copy_b128(&kbuf[p ^ 1][krow][chunk],
                                     khead + (size_t)(kb2 + krow) * HD + chunk);
      else           async_copy_b128(&vbuf[p ^ 1][vrow][chunk],
                                     vhead + (size_t)vrow * NTOK + kb2 + chunk);
    }

    // S = Q @ K^T from LDS: B column n = key row (contiguous in kbuf).
    v16h b0 = load_ab_f16(&kbuf[p][nn][0],      0, lo);
    v16h b1 = load_ab_f16(&kbuf[p][16 + nn][0], 0, lo);
    v8f z = {};
    v8f s0 = wmma16(qa, b0, z);
    v8f s1 = wmma16(qa, b1, z);

    // Online softmax.  C-layout row m = r + 8*hi lives in one 16-lane half,
    // so xor-masks 1/2/4/8 reduce exactly across the row.
#pragma unroll
    for (int r = 0; r < 8; ++r) {
      float a0 = s0[r] * SCALE, a1 = s1[r] * SCALE;
      float rmax = fmaxf(a0, a1);
#pragma unroll
      for (int m = 1; m <= 8; m <<= 1) rmax = fmaxf(rmax, __shfl_xor(rmax, m, 32));
      const float mnew  = fmaxf(mi[r], rmax);
      const float alpha = __expf(mi[r] - mnew);
      const float p0 = __expf(a0 - mnew);
      const float p1 = __expf(a1 - mnew);
      float rs = p0 + p1;
#pragma unroll
      for (int m = 1; m <= 8; m <<= 1) rs += __shfl_xor(rs, m, 32);
      li[r] = li[r] * alpha + rs;
      mi[r] = mnew;
      o0[r] *= alpha;
      o1[r] *= alpha;
      // C-layout -> LDS (row-major 16x32 P tile) for A-operand reload.
      const int mrow = r + 8 * hi;
      pbuf[mrow * 32 + nn]      = (_Float16)p0;
      pbuf[mrow * 32 + 16 + nn] = (_Float16)p1;
    }
    asm volatile("s_wait_dscnt 0x0" ::: "memory");  // P visible before reload

    // P as A operand; V rows (hd) over 32 local keys from LDS.
    v16h pa  = load_ab_f16(pbuf + (lane & 15) * 32, 0, lo);
    v16h vb0 = load_ab_f16(&vbuf[p][nn][0],      0, lo);
    v16h vb1 = load_ab_f16(&vbuf[p][16 + nn][0], 0, lo);
    o0 = wmma16(pa, vb0, o0);
    o1 = wmma16(pa, vb1, o1);
  }

  // Normalize and store pre-projection output [t, n, h*32 + d] as f16.
#pragma unroll
  for (int r = 0; r < 8; ++r) {
    const float inv = 1.0f / li[r];
    const int row = q0 + r + 8 * hi;
    const size_t ob = ((size_t)(t * NTOK + row)) * HIDDEN + h * HD;
    aout[ob + nn]      = (_Float16)(o0[r] * inv);
    aout[ob + 16 + nn] = (_Float16)(o1[r] * inv);
  }
}

// ---------------------------------------------------------------------------
// Kernel 3: out = attn @ Wo^T + bo, f16 A, f32 output.
// ---------------------------------------------------------------------------
__global__ __launch_bounds__(256) void out_proj_kernel(
    const _Float16* __restrict__ A, const float* __restrict__ W,
    const float* __restrict__ bias, float* __restrict__ out) {
  const int lane = threadIdx.x & 31;
  const int wave = threadIdx.x >> 5;
  const int tid  = blockIdx.x * 8 + wave;
  const int row_tile = tid >> 4;
  const int col_tile = tid & 15;
  const int lo = (lane < 16) ? 0 : 8;
  const int nn = lane & 15;
  const int hi = lane >> 4;

  const _Float16* arow = A + (size_t)(row_tile * 16 + nn) * HIDDEN;
  const float*    wrow = W + (size_t)(col_tile * 16 + nn) * HIDDEN;

  v8f acc = {};
#pragma unroll
  for (int kk = 0; kk < HIDDEN; kk += 32) {
    v16h a = load_ab_f16(arow, kk, lo);
    v16h b = load_ab_f32(wrow, kk, lo);
    acc = wmma16(a, b, acc);
  }

  const int col = col_tile * 16 + nn;
  const float bv = bias[col];
#pragma unroll
  for (int r = 0; r < 8; ++r) {
    const int row = row_tile * 16 + r + 8 * hi;
    out[(size_t)row * HIDDEN + col] = acc[r] + bv;   // coalesced b32 stores
  }
}

// ---------------------------------------------------------------------------
extern "C" void kernel_launch(void* const* d_in, const int* in_sizes, int n_in,
                              void* d_out, int out_size, void* d_ws, size_t ws_size,
                              hipStream_t stream) {
  const float* x   = (const float*)d_in[0];
  const float* ctx = (const float*)d_in[1];
  const float* Wq  = (const float*)d_in[2];
  const float* bq  = (const float*)d_in[3];
  const float* Wk  = (const float*)d_in[4];
  const float* bk  = (const float*)d_in[5];
  const float* Wv  = (const float*)d_in[6];
  const float* bv  = (const float*)d_in[7];
  const float* Wo  = (const float*)d_in[8];
  const float* bo  = (const float*)d_in[9];
  float* out = (float*)d_out;

  // Workspace: 4 x 4MB f16 buffers = 16 MB.
  const size_t elems = (size_t)ROWS * HIDDEN;   // 2M f16 each
  _Float16* qh = (_Float16*)d_ws;
  _Float16* kh = qh + elems;
  _Float16* vt = kh + elems;    // V stored transposed per head: [T,H,32,N]
  _Float16* ao = vt + elems;

  dim3 blk(256);
  proj_qkv_kernel<<<1024, blk, 0, stream>>>(x,   Wq, bq, qh, 0);
  proj_qkv_kernel<<<1024, blk, 0, stream>>>(ctx, Wk, bk, kh, 0);
  proj_qkv_kernel<<<1024, blk, 0, stream>>>(ctx, Wv, bv, vt, 1);
  attn_kernel    <<<512,  blk, 0, stream>>>(qh, kh, vt, ao);
  out_proj_kernel<<<1024, blk, 0, stream>>>(ao, Wo, bo, out);
}
